// EnKF_LSTM_60868276519435
// MI455X (gfx1250) — matrix operand order repaired and checked by reading, not compiled, not run on hardware
//
#include <hip/hip_runtime.h>
#include <hip/hip_bf16.h>

// ---------------------------------------------------------------------------
// EnKF + 2-layer LSTM for MI455X (gfx1250, wave32, WMMA).
// Dominant cost: 137 GFLOP of recurrent GEMM -> v_wmma_f32_16x16x32_bf16,
// weights resident in 320KB WGP LDS, staged via global_load_async_to_lds_b128
// (ASYNCcnt DMA). EnKF update/cov: per-(l,b) 128^3 WMMA GEMMs.
// Requires ws_size >= ~81 MB.
// ---------------------------------------------------------------------------

typedef __attribute__((ext_vector_type(16))) __bf16 v16bf;
typedef __attribute__((ext_vector_type(8)))  __bf16 v8bf;
typedef __attribute__((ext_vector_type(8)))  float  v8f;

#define LSN  2
#define BSN  64
#define TN   64
#define DN   128
#define NHN  128
#define NEN  128
#define NMN  64      // ensemble members (NE/2)
#define G4N  512     // 4*NH
#define KCN  256     // D + NH
#define MBLK 32      // batch rows per LSTM block

// ---- CDNA5 helpers -------------------------------------------------------
// Async DMA: global -> LDS, 16B per lane, tracked by ASYNCcnt.
// Generic LDS pointers carry the AS3 offset in their low 32 bits (ISA 10.2).
__device__ __forceinline__ void async_load_b128(void* lds, const void* g) {
  unsigned loff = (unsigned)(unsigned long long)(uintptr_t)lds;
  unsigned long long ga = (unsigned long long)(uintptr_t)g;
  asm volatile("global_load_async_to_lds_b128 %0, %1, off"
               :: "v"(loff), "v"(ga) : "memory");
}
__device__ __forceinline__ void async_wait0() {
#if __has_builtin(__builtin_amdgcn_s_wait_asynccnt)
  __builtin_amdgcn_s_wait_asynccnt(0);
#else
  asm volatile("s_wait_asynccnt 0x0" ::: "memory");
#endif
}
// gfx1250 hardware tanh (v_tanh_f32) if the builtin is exposed.
__device__ __forceinline__ float fast_tanh(float x) {
#if __has_builtin(__builtin_amdgcn_tanhf)
  return __builtin_amdgcn_tanhf(x);
#elif __has_builtin(__builtin_amdgcn_tanh_f32)
  return __builtin_amdgcn_tanh_f32(x);
#else
  return tanhf(x);
#endif
}
__device__ __forceinline__ float fast_sigmoid(float x) {
  return 1.f / (1.f + __expf(-x));
}

// ---- workspace layout (byte offsets) ----
static constexpr size_t WS_WCAT  = 0;                       // bf16 [2][512][256]
static constexpr size_t WS_BIAS  = 524288;                  // f32  [2][512]
static constexpr size_t WS_H0SEQ = 1048576;                 // bf16 [64][64][64][128]
static constexpr size_t WS_PRED  = WS_H0SEQ + 67108864;     // f32  [2][64][128][128]
static constexpr size_t WS_HXI   = WS_PRED + 8388608;       // f32  [64][128]
static constexpr size_t WS_HA    = WS_HXI + 32768;          // f32  [64][128]
static constexpr size_t WS_INNOV = WS_HA + 32768;           // f32  [64][128]
static constexpr size_t WS_K2    = WS_INNOV + 32768;        // f32  [64][128]
static constexpr size_t WS_PMAT  = WS_K2 + 32768;           // f32  [64][64]
static constexpr size_t WS_PINV  = WS_PMAT + 16384;         // f32  [64][64]
static constexpr size_t WS_WMAT  = WS_PINV + 16384;         // f32  [128][128]
static constexpr size_t WS_ABAR  = WS_WMAT + 65536;         // f32  [2][64][128]
static constexpr size_t WS_AN    = WS_ABAR + 65536;         // bf16 [2][64][128][128]

// ---- d_out layout (float offsets) ----
static constexpr size_t OUT_FO   = 0;          // filter_out (64)
static constexpr size_t OUT_FCOV = 64;         // filter_cov (4096)
static constexpr size_t OUT_X    = 4160;       // X_new (2097152)
static constexpr size_t OUT_COV  = 2101312;    // Cov_new (2097152)
static constexpr size_t OUT_LL   = 4198464;    // ll (1)

// ===========================================================================
// Weight prep: fuse [W_ih | W_hh] -> bf16, bias = b_ih + b_hh
// ===========================================================================
__global__ void prep_kernel(const float* __restrict__ Wih,
                            const float* __restrict__ Whh,
                            const float* __restrict__ bih,
                            const float* __restrict__ bhh,
                            __bf16* __restrict__ wcat,
                            float* __restrict__ bias) {
  const int stride = gridDim.x * blockDim.x;
  for (int i = blockIdx.x * blockDim.x + threadIdx.x; i < LSN * G4N * KCN; i += stride) {
    int k = i & (KCN - 1);
    int g = (i >> 8) & (G4N - 1);
    int l = i >> 17;
    float v = (k < DN) ? Wih[((size_t)l * G4N + g) * DN + k]
                       : Whh[((size_t)l * G4N + g) * NHN + (k - DN)];
    wcat[i] = (__bf16)v;
  }
  for (int i = blockIdx.x * blockDim.x + threadIdx.x; i < LSN * G4N; i += stride)
    bias[i] = bih[i] + bhh[i];
}

// ===========================================================================
// Fused 2-layer LSTM. Block = (member, batch-half). 8 waves / 256 threads.
// Layer weights (512x256 bf16 = 256KB) resident in LDS across T; staged by
// async DMA that overlaps the h/c initial-state gathers.
// ===========================================================================
__global__ __launch_bounds__(256) void lstm_kernel(
    const float* __restrict__ x,      // (BS,T,D)
    const float* __restrict__ uhi,    // (LS,BS,NH,NE)
    const __bf16* __restrict__ wcat,  // (LS,G4,KC)
    const float* __restrict__ bias,   // (LS,G4)
    __bf16* __restrict__ h0seq,       // (NM,T,BS,NH)
    float* __restrict__ pred)         // (LS,BS,NH,NE)
{
  extern __shared__ char smem[];
  __bf16* lds_w  = (__bf16*)smem;                // 512*256 bf16 = 262144 B
  __bf16* lds_xh = (__bf16*)(smem + 262144);     // 32*256 bf16  = 16384 B

  const int tid   = threadIdx.x;
  const int wave  = tid >> 5;
  const int lane  = tid & 31;
  const int lhalf = lane >> 4;
  const int lc    = lane & 15;
  const int e     = blockIdx.x >> 1;             // ensemble member
  const int B0    = (blockIdx.x & 1) * MBLK;     // batch row base
  const int jcol  = wave * 16 + lc;              // owned hidden column

  float c_reg[2][8];

  for (int l = 0; l < LSN; ++l) {
    // --- stage layer weights into LDS via async DMA (overlaps init below) ---
    {
      const uint4* src = (const uint4*)(wcat + (size_t)l * G4N * KCN);
      for (int i = tid; i < 16384; i += 256)
        async_load_b128((uint4*)lds_w + i, src + i);
    }
    // --- init h into lds_xh cols D..D+127; c into registers ---
    for (int i = tid; i < MBLK * NHN; i += 256) {
      int bl = i >> 7, j = i & 127;
      int b = B0 + bl;
      lds_xh[bl * KCN + DN + j] =
          (__bf16)uhi[(((size_t)l * BSN + b) * NHN + j) * NEN + e];
    }
#pragma unroll
    for (int mt = 0; mt < 2; ++mt)
#pragma unroll
      for (int v = 0; v < 8; ++v) {
        int b = B0 + mt * 16 + v + lhalf * 8;
        c_reg[mt][v] = uhi[(((size_t)l * BSN + b) * NHN + jcol) * NEN + (NMN + e)];
      }
    const float bi  = bias[l * G4N + 0 * NHN + jcol];
    const float bff = bias[l * G4N + 1 * NHN + jcol];
    const float bg  = bias[l * G4N + 2 * NHN + jcol];
    const float bo  = bias[l * G4N + 3 * NHN + jcol];
    async_wait0();
    __syncthreads();

    for (int t = 0; t < TN; ++t) {
      // --- stage step input into lds_xh cols 0..127 ---
      if (l == 0) {
        for (int i = tid; i < (MBLK * DN) / 4; i += 256) {
          int bl = i >> 5, d4 = i & 31;
          float4 vx = ((const float4*)(x + ((size_t)(B0 + bl) * TN + t) * DN))[d4];
          int c0 = d4 * 4;
          lds_xh[bl * KCN + c0 + 0] = (__bf16)vx.x;
          lds_xh[bl * KCN + c0 + 1] = (__bf16)vx.y;
          lds_xh[bl * KCN + c0 + 2] = (__bf16)vx.z;
          lds_xh[bl * KCN + c0 + 3] = (__bf16)vx.w;
        }
      } else {
        // bf16 h_{t} of layer 0: pure byte copy -> async DMA to LDS
        for (int i = tid; i < (MBLK * NHN) / 8; i += 256) {
          int bl = i >> 4, c8 = (i & 15) * 8;
          async_load_b128(lds_xh + bl * KCN + c8,
                          h0seq + ((((size_t)e * TN + t) * BSN) + B0 + bl) * NHN + c8);
        }
        async_wait0();
      }
      __syncthreads();

      // --- GEMM: g(32 x 512) = [x|h](32 x 256) @ Wcat^T via WMMA bf16 ---
      v8f acc[2][4] = {};
      for (int k0 = 0; k0 < KCN; k0 += 32) {
        v16bf afrag[2];
#pragma unroll
        for (int mt = 0; mt < 2; ++mt) {
          const __bf16* ap = lds_xh + (mt * 16 + lc) * KCN + k0 + lhalf * 16;
          v8bf alo = *(const v8bf*)ap;
          v8bf ahi = *(const v8bf*)(ap + 8);
#pragma unroll
          for (int q = 0; q < 8; ++q) { afrag[mt][q] = alo[q]; afrag[mt][8 + q] = ahi[q]; }
        }
#pragma unroll
        for (int g = 0; g < 4; ++g) {
          int nrow = (wave + 8 * g) * 16 + lc;   // Wcat row (gate-fused tiling)
          const __bf16* bp = lds_w + nrow * KCN + k0 + lhalf * 16;
          v8bf blo = *(const v8bf*)bp;
          v8bf bhi = *(const v8bf*)(bp + 8);
          v16bf bfrag;
#pragma unroll
          for (int q = 0; q < 8; ++q) { bfrag[q] = blo[q]; bfrag[8 + q] = bhi[q]; }
#pragma unroll
          for (int mt = 0; mt < 2; ++mt) {
            acc[mt][g] = __builtin_amdgcn_wmma_f32_16x16x32_bf16(
                false, afrag[mt], false, bfrag, (short)0, acc[mt][g], false, false);
          }
        }
      }
      __syncthreads();  // all waves done reading lds_xh

      // --- gate fusion + state update; h written back into lds_xh ---
#pragma unroll
      for (int mt = 0; mt < 2; ++mt) {
#pragma unroll
        for (int v = 0; v < 8; ++v) {
          int bl = mt * 16 + v + lhalf * 8;
          float gi = acc[mt][0][v] + bi;
          float gf = acc[mt][1][v] + bff;
          float gg = acc[mt][2][v] + bg;
          float go = acc[mt][3][v] + bo;
          float cn = fast_sigmoid(gf) * c_reg[mt][v] + fast_sigmoid(gi) * fast_tanh(gg);
          c_reg[mt][v] = cn;
          lds_xh[bl * KCN + DN + jcol] = (__bf16)(fast_sigmoid(go) * fast_tanh(cn));
        }
      }
      __syncthreads();

      // --- layer 0: spill h_t for layer-1 consumption ---
      if (l == 0) {
        for (int i = tid; i < (MBLK * NHN) / 8; i += 256) {
          int bl = i >> 4, c8 = (i & 15) * 8;
          uint4 vv = *(const uint4*)(lds_xh + bl * KCN + DN + c8);
          *(uint4*)(h0seq + ((((size_t)e * TN + t) * BSN) + B0 + bl) * NHN + c8) = vv;
        }
      }
    }

    // --- write final hT, cT into predicted ensemble ---
    for (int i = tid; i < MBLK * NHN; i += 256) {
      int bl = i >> 7, j = i & 127;
      int b = B0 + bl;
      pred[(((size_t)l * BSN + b) * NHN + j) * NEN + e] =
          (float)lds_xh[bl * KCN + DN + j];
    }
#pragma unroll
    for (int mt = 0; mt < 2; ++mt)
#pragma unroll
      for (int v = 0; v < 8; ++v) {
        int b = B0 + mt * 16 + v + lhalf * 8;
        pred[(((size_t)l * BSN + b) * NHN + jcol) * NEN + (NMN + e)] = c_reg[mt][v];
      }
    __syncthreads();
  }
}

// ===========================================================================
// pred += |q| * noise_h / |e| * noise_c  (builds uhi_pred in place)
// ===========================================================================
__global__ void noise_kernel(float* __restrict__ pred,
                             const float* __restrict__ noise_h,
                             const float* __restrict__ noise_c,
                             const float* __restrict__ q,
                             const float* __restrict__ ev) {
  int idx = blockIdx.x * blockDim.x + threadIdx.x;   // exact 2^21 threads
  int N = idx & 127;
  int n = (idx >> 7) & 127;
  int b = (idx >> 14) & 63;
  int l = idx >> 20;
  float add;
  if (N < NMN)
    add = fabsf(q[0]) * noise_h[(((size_t)N * LSN + l) * BSN + b) * NHN + n];
  else
    add = fabsf(ev[0]) * noise_c[(((size_t)(N - NMN) * LSN + l) * BSN + b) * NHN + n];
  pred[idx] += add;
}

// hxi[b][N] = sum_n pred[0,b,n,N] * H[n]
__global__ void hxi_kernel(const float* __restrict__ pred,
                           const float* __restrict__ H,
                           float* __restrict__ hxi) {
  int i = blockIdx.x * blockDim.x + threadIdx.x;  // 8192
  int b = i >> 7, N = i & 127;
  float s = 0.f;
  for (int n = 0; n < NHN; ++n) s += pred[((size_t)b * NHN + n) * NEN + N] * H[n];
  hxi[i] = s;
}

// HA = hxi - mean_N ; innov = y - hxi
__global__ void center_kernel(const float* __restrict__ hxi,
                              const float* __restrict__ y,
                              float* __restrict__ HA,
                              float* __restrict__ innov) {
  __shared__ float red[128];
  int b = blockIdx.x, N = threadIdx.x;
  float v = hxi[b * NEN + N];
  red[N] = v;
  __syncthreads();
  for (int s = 64; s > 0; s >>= 1) {
    if (N < s) red[N] += red[N + s];
    __syncthreads();
  }
  float mean = red[0] * (1.f / NEN);
  HA[b * NEN + N] = v - mean;
  innov[b * NEN + N] = y[b] - v;
}

// Abar[l,b,n] = mean_N pred
__global__ void abar_kernel(const float* __restrict__ pred, float* __restrict__ abar) {
  int i = blockIdx.x * blockDim.x + threadIdx.x;  // 16384
  int lb = i >> 7, n = i & 127;
  float s = 0.f;
  const float* p = pred + (size_t)lb * NHN * NEN + (size_t)n * NEN;
  for (int N = 0; N < NEN; ++N) s += p[N];
  abar[i] = s * (1.f / NEN);
}

// P = HA@HA^T/(N-1) + r^2 I
__global__ void pmat_kernel(const float* __restrict__ HA,
                            const float* __restrict__ r,
                            float* __restrict__ P) {
  int i = blockIdx.x * blockDim.x + threadIdx.x;  // 4096
  int c = i >> 6, d = i & 63;
  float s = 0.f;
  for (int N = 0; N < NEN; ++N) s += HA[c * NEN + N] * HA[d * NEN + N];
  float v = s * (1.f / (NEN - 1));
  if (c == d) v += r[0] * r[0];
  P[i] = v;
}

// Gauss-Jordan inverse of 64x64 SPD matrix, single block.
__global__ void inv_kernel(const float* __restrict__ P, float* __restrict__ Pinv) {
  extern __shared__ char smem[];
  float* aug  = (float*)smem;            // 64 x 128
  float* fcol = (float*)(smem + 32768);  // 64
  const int tid = threadIdx.x;
  for (int i = tid; i < 64 * 128; i += 256) {
    int row = i >> 7, col = i & 127;
    aug[i] = (col < 64) ? P[row * 64 + col] : ((col - 64 == row) ? 1.f : 0.f);
  }
  __syncthreads();
  for (int k = 0; k < 64; ++k) {
    float ipiv = 1.f / aug[k * 128 + k];
    __syncthreads();
    if (tid < 128) aug[k * 128 + tid] *= ipiv;
    if (tid >= 128 && tid < 192) {
      int i = tid - 128;
      fcol[i] = (i == k) ? 0.f : aug[i * 128 + k];
    }
    __syncthreads();
    for (int idx = tid; idx < 64 * 128; idx += 256) {
      int i = idx >> 7, j = idx & 127;
      if (i != k) aug[idx] -= fcol[i] * aug[k * 128 + j];
    }
    __syncthreads();
  }
  for (int i = tid; i < 64 * 64; i += 256) {
    int row = i >> 6, col = i & 63;
    Pinv[i] = aug[row * 128 + 64 + col];
  }
}

// K2 = Pinv @ innov
__global__ void k2_kernel(const float* __restrict__ Pinv,
                          const float* __restrict__ innov,
                          float* __restrict__ K2) {
  int i = blockIdx.x * blockDim.x + threadIdx.x;  // 8192
  int c = i >> 7, M = i & 127;
  float s = 0.f;
  for (int d = 0; d < 64; ++d) s += Pinv[c * 64 + d] * innov[d * NEN + M];
  K2[i] = s;
}

// wmat[N][M] = (1/(NE-1)) * sum_c HA[c,N] * K2[c,M]
__global__ void wmat_kernel(const float* __restrict__ HA,
                            const float* __restrict__ K2,
                            float* __restrict__ wmat) {
  int i = blockIdx.x * blockDim.x + threadIdx.x;  // 16384
  int N = i >> 7, M = i & 127;
  float s = 0.f;
  for (int c = 0; c < 64; ++c) s += HA[c * NEN + N] * K2[c * NEN + M];
  wmat[i] = s * (1.f / (NEN - 1));
}

// ===========================================================================
// X_new[l,b] = pred + (pred - Abar) @ wmat  (128x128x128 WMMA per block)
// Also emits An = X_new - rowmean(X_new) as bf16 for the Cov kernel.
// ===========================================================================
__global__ __launch_bounds__(256) void update_kernel(
    const float* __restrict__ pred,
    const float* __restrict__ abar,
    const float* __restrict__ wmat,
    float* __restrict__ xnew,
    __bf16* __restrict__ an_ws) {
  extern __shared__ char smem[];
  __bf16* a_bf = (__bf16*)smem;             // 128x128
  __bf16* w_t  = (__bf16*)(smem + 32768);   // wmat transposed [M][N]
  float*  rsum = (float*)(smem + 65536);    // 128
  const int tid = threadIdx.x, wave = tid >> 5, lane = tid & 31;
  const int lhalf = lane >> 4, lc = lane & 15;
  const int lb = blockIdx.x;
  const size_t base = (size_t)lb * NHN * NEN;

  for (int i = tid; i < NHN * NEN; i += 256) {
    int n = i >> 7;
    a_bf[i] = (__bf16)(pred[base + i] - abar[lb * NHN + n]);
  }
  for (int i = tid; i < NEN * NEN; i += 256) {
    int m = i >> 7, k = i & 127;
    w_t[i] = (__bf16)wmat[k * NEN + m];
  }
  if (tid < 128) rsum[tid] = 0.f;
  __syncthreads();

  v8f acc[8] = {};
  const int mrow = wave * 16 + lc;
  for (int k0 = 0; k0 < NEN; k0 += 32) {
    v16bf afrag;
    {
      const __bf16* ap = a_bf + mrow * NEN + k0 + lhalf * 16;
      v8bf lo = *(const v8bf*)ap;
      v8bf hi = *(const v8bf*)(ap + 8);
#pragma unroll
      for (int q = 0; q < 8; ++q) { afrag[q] = lo[q]; afrag[8 + q] = hi[q]; }
    }
#pragma unroll
    for (int nt = 0; nt < 8; ++nt) {
      const __bf16* bp = w_t + (nt * 16 + lc) * NEN + k0 + lhalf * 16;
      v8bf lo = *(const v8bf*)bp;
      v8bf hi = *(const v8bf*)(bp + 8);
      v16bf bfrag;
#pragma unroll
      for (int q = 0; q < 8; ++q) { bfrag[q] = lo[q]; bfrag[8 + q] = hi[q]; }
      acc[nt] = __builtin_amdgcn_wmma_f32_16x16x32_bf16(
          false, afrag, false, bfrag, (short)0, acc[nt], false, false);
    }
  }
  __syncthreads();

  float xv[8][8];
#pragma unroll
  for (int nt = 0; nt < 8; ++nt)
#pragma unroll
    for (int v = 0; v < 8; ++v) {
      int n = wave * 16 + v + lhalf * 8;
      int col = nt * 16 + lc;
      float val = pred[base + (size_t)n * NEN + col] + acc[nt][v];
      xnew[base + (size_t)n * NEN + col] = val;
      xv[nt][v] = val;
    }
#pragma unroll
  for (int v = 0; v < 8; ++v) {
    float s = 0.f;
#pragma unroll
    for (int nt = 0; nt < 8; ++nt) s += xv[nt][v];
    atomicAdd(&rsum[wave * 16 + v + lhalf * 8], s);
  }
  __syncthreads();
#pragma unroll
  for (int nt = 0; nt < 8; ++nt)
#pragma unroll
    for (int v = 0; v < 8; ++v) {
      int n = wave * 16 + v + lhalf * 8;
      int col = nt * 16 + lc;
      an_ws[base + (size_t)n * NEN + col] =
          (__bf16)(xv[nt][v] - rsum[n] * (1.f / NEN));
    }
}

// ===========================================================================
// Cov_new[l,b] = An @ An^T / (NE-1)    (128x128x128 WMMA per block)
// ===========================================================================
__global__ __launch_bounds__(256) void cov_kernel(const __bf16* __restrict__ an_ws,
                                                  float* __restrict__ cov) {
  extern __shared__ char smem[];
  __bf16* an = (__bf16*)smem;  // 128x128 bf16
  const int tid = threadIdx.x, wave = tid >> 5, lane = tid & 31;
  const int lhalf = lane >> 4, lc = lane & 15;
  const int lb = blockIdx.x;
  const size_t base = (size_t)lb * NHN * NEN;

  {
    const uint4* src = (const uint4*)(an_ws + base);
    for (int i = tid; i < 2048; i += 256)
      async_load_b128((uint4*)an + i, src + i);
    async_wait0();
  }
  __syncthreads();

  v8f acc[8] = {};
  const int mrow = wave * 16 + lc;
  for (int k0 = 0; k0 < NEN; k0 += 32) {
    v16bf afrag;
    {
      const __bf16* ap = an + mrow * NEN + k0 + lhalf * 16;
      v8bf lo = *(const v8bf*)ap;
      v8bf hi = *(const v8bf*)(ap + 8);
#pragma unroll
      for (int q = 0; q < 8; ++q) { afrag[q] = lo[q]; afrag[8 + q] = hi[q]; }
    }
#pragma unroll
    for (int nt = 0; nt < 8; ++nt) {
      const __bf16* bp = an + (nt * 16 + lc) * NEN + k0 + lhalf * 16;
      v8bf lo = *(const v8bf*)bp;
      v8bf hi = *(const v8bf*)(bp + 8);
      v16bf bfrag;
#pragma unroll
      for (int q = 0; q < 8; ++q) { bfrag[q] = lo[q]; bfrag[8 + q] = hi[q]; }
      acc[nt] = __builtin_amdgcn_wmma_f32_16x16x32_bf16(
          false, afrag, false, bfrag, (short)0, acc[nt], false, false);
    }
  }
#pragma unroll
  for (int nt = 0; nt < 8; ++nt)
#pragma unroll
    for (int v = 0; v < 8; ++v) {
      int n = wave * 16 + v + lhalf * 8;
      int m = nt * 16 + lc;
      cov[base + (size_t)n * NEN + m] = acc[nt][v] * (1.f / (NEN - 1));
    }
}

// ===========================================================================
// Filter head: Y = H^T X_new[0], filter_out, filter_cov, log-likelihood
// (single block; 64x64 Cholesky + forward solve)
// ===========================================================================
__global__ __launch_bounds__(256) void finalize_kernel(
    const float* __restrict__ xnew,   // X_new region of d_out
    const float* __restrict__ H,
    const float* __restrict__ y,
    const float* __restrict__ r,
    float* __restrict__ out_fo,
    float* __restrict__ out_fcov,
    float* __restrict__ out_ll) {
  extern __shared__ char smem[];
  float* Ys   = (float*)smem;                  // 64x128
  float* fcs  = (float*)(smem + 32768);        // 64x64
  float* Hs   = (float*)(smem + 49152);        // 128
  float* fos  = (float*)(smem + 49664);        // 64
  float* inns = (float*)(smem + 49920);        // 64
  const int tid = threadIdx.x;

  if (tid < 128) Hs[tid] = H[tid];
  __syncthreads();

  for (int i = tid; i < BSN * NEN; i += 256) {
    int b = i >> 7, N = i & 127;
    float s = 0.f;
    for (int n = 0; n < NHN; ++n) s += xnew[((size_t)b * NHN + n) * NEN + N] * Hs[n];
    Ys[i] = s;
  }
  __syncthreads();
  if (tid < 64) {
    float s = 0.f;
    for (int N = 0; N < NEN; ++N) s += Ys[tid * NEN + N];
    float mean = s * (1.f / NEN);
    fos[tid] = mean;
    inns[tid] = y[tid] - mean;
    out_fo[tid] = mean;
  }
  __syncthreads();
  for (int i = tid; i < BSN * NEN; i += 256) Ys[i] -= fos[i >> 7];
  __syncthreads();
  const float r2 = r[0] * r[0];
  for (int i = tid; i < 64 * 64; i += 256) {
    int c = i >> 6, d = i & 63;
    float s = 0.f;
    for (int N = 0; N < NEN; ++N) s += Ys[c * NEN + N] * Ys[d * NEN + N];
    float v = s * (1.f / (NEN - 1));
    if (c == d) v += r2;
    fcs[i] = v;
    out_fcov[i] = v;
  }
  __syncthreads();

  // Cholesky (lower) in place
  for (int k = 0; k < 64; ++k) {
    if (tid == 0) fcs[k * 64 + k] = sqrtf(fcs[k * 64 + k]);
    __syncthreads();
    if (tid > k && tid < 64) fcs[tid * 64 + k] /= fcs[k * 64 + k];
    __syncthreads();
    for (int idx = tid; idx < 64 * 64; idx += 256) {
      int i = idx >> 6, j = idx & 63;
      if (i > k && j > k && j <= i)
        fcs[i * 64 + j] -= fcs[i * 64 + k] * fcs[j * 64 + k];
    }
    __syncthreads();
  }
  if (tid == 0) {
    float logdet = 0.f;
    for (int i = 0; i < 64; ++i) logdet += logf(fcs[i * 65]);
    logdet *= 2.f;
    float z[64];
    for (int i = 0; i < 64; ++i) {
      float s = inns[i];
      for (int j = 0; j < i; ++j) s -= fcs[i * 64 + j] * z[j];
      z[i] = s / fcs[i * 65];
    }
    float quad = 0.f;
    for (int i = 0; i < 64; ++i) quad += z[i] * z[i];
    out_ll[0] = -0.5f * logdet - 0.5f * quad;
  }
}

// ===========================================================================
extern "C" void kernel_launch(void* const* d_in, const int* in_sizes, int n_in,
                              void* d_out, int out_size, void* d_ws, size_t ws_size,
                              hipStream_t stream) {
  const float* x       = (const float*)d_in[0];
  const float* y       = (const float*)d_in[1];
  const float* uhi     = (const float*)d_in[2];
  const float* Wih     = (const float*)d_in[3];
  const float* Whh     = (const float*)d_in[4];
  const float* bih     = (const float*)d_in[5];
  const float* bhh     = (const float*)d_in[6];
  const float* H       = (const float*)d_in[7];
  const float* q       = (const float*)d_in[8];
  const float* ev      = (const float*)d_in[9];
  const float* r       = (const float*)d_in[10];
  const float* noise_h = (const float*)d_in[11];
  const float* noise_c = (const float*)d_in[12];

  char* ws = (char*)d_ws;
  __bf16* wcat   = (__bf16*)(ws + WS_WCAT);
  float*  bias   = (float*)(ws + WS_BIAS);
  __bf16* h0seq  = (__bf16*)(ws + WS_H0SEQ);
  float*  pred   = (float*)(ws + WS_PRED);
  float*  hxi    = (float*)(ws + WS_HXI);
  float*  HA     = (float*)(ws + WS_HA);
  float*  innov  = (float*)(ws + WS_INNOV);
  float*  K2     = (float*)(ws + WS_K2);
  float*  P      = (float*)(ws + WS_PMAT);
  float*  Pinv   = (float*)(ws + WS_PINV);
  float*  wmat   = (float*)(ws + WS_WMAT);
  float*  abar   = (float*)(ws + WS_ABAR);
  __bf16* an_ws  = (__bf16*)(ws + WS_AN);

  float* out = (float*)d_out;

  (void)hipFuncSetAttribute(reinterpret_cast<const void*>(&lstm_kernel),
                            hipFuncAttributeMaxDynamicSharedMemorySize, 278528);
  (void)hipFuncSetAttribute(reinterpret_cast<const void*>(&update_kernel),
                            hipFuncAttributeMaxDynamicSharedMemorySize, 66048);

  prep_kernel<<<256, 256, 0, stream>>>(Wih, Whh, bih, bhh, wcat, bias);
  lstm_kernel<<<128, 256, 278528, stream>>>(x, uhi, wcat, bias, h0seq, pred);
  noise_kernel<<<8192, 256, 0, stream>>>(pred, noise_h, noise_c, q, ev);
  hxi_kernel<<<32, 256, 0, stream>>>(pred, H, hxi);
  center_kernel<<<64, 128, 0, stream>>>(hxi, y, HA, innov);
  abar_kernel<<<64, 256, 0, stream>>>(pred, abar);
  pmat_kernel<<<16, 256, 0, stream>>>(HA, r, P);
  inv_kernel<<<1, 256, 33024, stream>>>(P, Pinv);
  k2_kernel<<<32, 256, 0, stream>>>(Pinv, innov, K2);
  wmat_kernel<<<64, 256, 0, stream>>>(HA, K2, wmat);
  update_kernel<<<128, 256, 66048, stream>>>(pred, abar, wmat, out + OUT_X, an_ws);
  cov_kernel<<<128, 256, 32768, stream>>>(an_ws, out + OUT_COV);
  finalize_kernel<<<1, 256, 50176, stream>>>(out + OUT_X, H, y, r,
                                             out + OUT_FO, out + OUT_FCOV, out + OUT_LL);
}
